// hConvGRUCell_79860621902136
// MI455X (gfx1250) — matrix operand order repaired and checked by reading, not compiled, not run on hardware
//
#include <hip/hip_runtime.h>

typedef __bf16 bf16;
typedef __attribute__((ext_vector_type(8)))  __bf16 bfx8;
typedef __attribute__((ext_vector_type(16))) __bf16 bfx16;
typedef __attribute__((ext_vector_type(8)))  float  floatx8;
typedef __attribute__((ext_vector_type(4)))  unsigned int u32x4;
typedef int gvec4i __attribute__((vector_size(16)));   // matches builtin pointee type

#define CCH   128
#define HWPX  4096   // 64*64
#define NB    8
#define NTOT  (NB * CCH * HWPX)   // 4,194,304
#define PITCH 72     // bf16 elements per LDS row (64 data + 8 pad), 144B (16B aligned)

// CDNA5 async global->LDS copy (ASYNCcnt-tracked). Guarded: fall back to
// synchronous staging if the builtin is absent on this toolchain.
#if defined(__has_builtin)
#  if __has_builtin(__builtin_amdgcn_global_load_async_to_lds_b128)
#    define HAVE_ASYNC 1
#  endif
#endif
#ifndef HAVE_ASYNC
#  define HAVE_ASYNC 0
#endif

#if HAVE_ASYNC
#define ASYNC_CP16(dst_lds, src_gbl)                                          \
    __builtin_amdgcn_global_load_async_to_lds_b128(                           \
        (__attribute__((address_space(1))) gvec4i*)(src_gbl),                 \
        (__attribute__((address_space(3))) gvec4i*)(dst_lds), 0, 0)
#define ASYNC_WAIT() asm volatile("s_wait_asynccnt 0x0" ::: "memory")
#else
#define ASYNC_CP16(dst_lds, src_gbl)                                          \
    (*(u32x4*)(dst_lds) = *(const u32x4*)(src_gbl))
#define ASYNC_WAIT()
#endif

// ---------------------------------------------------------------------------
// Weight repack: (O,C,7,7) f32 -> [tap][o][c] bf16 (contiguous per-tap slices)
// ---------------------------------------------------------------------------
__global__ void k_pack_wconv(const float* __restrict__ w, bf16* __restrict__ wpk) {
    int i = blockIdx.x * 256 + threadIdx.x;        // 128*128*49 = 802816
    if (i >= 128 * 128 * 49) return;
    int tap = i % 49;
    int rest = i / 49;
    int c = rest & 127;
    int o = rest >> 7;
    wpk[(tap * 128 + o) * 128 + c] = (bf16)w[i];
}

__global__ void k_f2bf(const float* __restrict__ x, bf16* __restrict__ y, int n) {
    int i = blockIdx.x * 256 + threadIdx.x;
    if (i < n) y[i] = (bf16)x[i];
}

// ---------------------------------------------------------------------------
// Attention gate: att = sigmoid(conv7(input, a_w) + a_w_b + conv7(exc, a_u) + a_u_b)
// ---------------------------------------------------------------------------
__global__ __launch_bounds__(256) void k_attgate(
    const float* __restrict__ inp, const float* __restrict__ exc,
    const float* __restrict__ aw,  const float* __restrict__ au,
    const float* __restrict__ awb, const float* __restrict__ aub,
    float* __restrict__ att)
{
    const int b  = blockIdx.x >> 4;
    const int y0 = (blockIdx.x & 15) * 4;
    const int t  = threadIdx.x;
    const int yl = t >> 6, x = t & 63;

    __shared__ float wA[32 * 49], wU[32 * 49];
    __shared__ float sI[10 * 64], sE[10 * 64];

    float acc = awb[0] + aub[0];

    for (int c0 = 0; c0 < 128; c0 += 32) {
        __syncthreads();
        for (int q = t; q < 32 * 49; q += 256) {
            int cl = q / 49, tap = q % 49;
            wA[q] = aw[(c0 + cl) * 49 + tap];
            wU[q] = au[(c0 + cl) * 49 + tap];
        }
        for (int c = c0; c < c0 + 32; ++c) {
            __syncthreads();
            const float* ib = inp + (((size_t)b * 128 + c) << 12);
            const float* eb = exc + (((size_t)b * 128 + c) << 12);
            for (int q = t; q < 640; q += 256) {
                int rr = q >> 6, xx = q & 63;
                int gy = y0 - 3 + rr;
                float vi = 0.f, ve = 0.f;
                if (gy >= 0 && gy < 64) { vi = ib[gy * 64 + xx]; ve = eb[gy * 64 + xx]; }
                sI[q] = vi; sE[q] = ve;
            }
            __syncthreads();
            const float* wa = &wA[(c - c0) * 49];
            const float* wu = &wU[(c - c0) * 49];
            #pragma unroll
            for (int r = 0; r < 7; ++r) {
                const int row = (yl + r) * 64;
                #pragma unroll
                for (int s = 0; s < 7; ++s) {
                    int sx = x + s - 3;
                    if (sx >= 0 && sx < 64) {
                        acc = fmaf(wa[r * 7 + s], sI[row + sx], acc);
                        acc = fmaf(wu[r * 7 + s], sE[row + sx], acc);
                    }
                }
            }
        }
    }
    att[((size_t)b << 12) + (y0 + yl) * 64 + x] = 1.f / (1.f + __expf(-acc));
}

// ---------------------------------------------------------------------------
// Pack activations NCHW f32 -> NHWC bf16 (4 outputs; xg = excitation*att_gate)
// ---------------------------------------------------------------------------
__global__ void k_pack_acts(const float* __restrict__ inp, const float* __restrict__ inh,
                            const float* __restrict__ exc, const float* __restrict__ att,
                            bf16* __restrict__ inp_bf, bf16* __restrict__ inh_bf,
                            bf16* __restrict__ exc_bf, bf16* __restrict__ xg_bf) {
    int i = blockIdx.x * 256 + threadIdx.x;
    if (i >= NTOT) return;
    int p = i & 4095;
    int c = (i >> 12) & 127;
    int b = i >> 19;
    size_t o = ((((size_t)b << 12) + p) << 7) + c;   // NHWC
    float a = att[((size_t)b << 12) + p];
    inp_bf[o] = (bf16)inp[i];
    inh_bf[o] = (bf16)inh[i];
    float e = exc[i];
    exc_bf[o] = (bf16)e;
    xg_bf[o]  = (bf16)(e * a);
}

// ---------------------------------------------------------------------------
// 7x7 conv as tap-decomposed implicit GEMM on WMMA bf16 (f32 accumulate).
// Double-buffered LDS pipeline with async global->LDS staging:
// stage s+1 streams into buf^1 while WMMAs consume buf.
// Block = 256 threads = 8 waves; tile M=128 x N=128 pixels; grid = B*32 = 256.
// ---------------------------------------------------------------------------
__global__ __launch_bounds__(256) void k_conv7_wmma(
    const bf16* __restrict__ Xt, const bf16* __restrict__ Wpk, float* __restrict__ Y)
{
    const int b    = blockIdx.x >> 5;
    const int n0   = (blockIdx.x & 31) << 7;
    const int t    = threadIdx.x;
    const int lane = t & 31, wid = t >> 5;

    __shared__ bf16 As[2][128 * PITCH];
    __shared__ bf16 Bs[2][128 * PITCH];

    const floatx8 zero = {0.f,0.f,0.f,0.f,0.f,0.f,0.f,0.f};
    floatx8 acc[8];
    #pragma unroll
    for (int i = 0; i < 8; ++i) acc[i] = zero;

    const int arow = (wid << 4) + (lane & 15);
    const int kb8  = (lane >> 4) << 3;   // A-frag K base within 32-chunk: 0 or 8
    const int kb16 = (lane >> 4) << 4;   // unused; B-frag K base is kb8 (0 or 8), same as A
    const int pxl  = lane & 15;

    // stage st (0..97): tap = st>>1, 64-channel chunk = st&1
    auto load_stage = [&](int st, int buf) {
        const int tap = st >> 1;
        const int c0  = (st & 1) << 6;
        const int r = tap / 7, s = tap % 7;
        const bf16* wsl = Wpk + (size_t)tap * 16384 + c0;
        // A: 128 o-rows x 64 c
        for (int q = t; q < 1024; q += 256) {
            int o = q >> 3, cc = (q & 7) << 3;
            ASYNC_CP16(&As[buf][o * PITCH + cc], &wsl[o * 128 + cc]);
        }
        // B (px-major): 128 px-rows x 64 c, spatially shifted by tap
        for (int q = t; q < 1024; q += 256) {
            int px = q >> 3, cc = (q & 7) << 3;
            int p  = n0 + px;
            int sy = (p >> 6) + r - 3;
            int sx = (p & 63) + s - 3;
            if (sy >= 0 && sy < 64 && sx >= 0 && sx < 64) {
                ASYNC_CP16(&Bs[buf][px * PITCH + cc],
                           &Xt[((((size_t)b << 12) + sy * 64 + sx) << 7) + c0 + cc]);
            } else {
                u32x4 z = {0u, 0u, 0u, 0u};
                *(u32x4*)&Bs[buf][px * PITCH + cc] = z;
            }
        }
    };

    load_stage(0, 0);
    ASYNC_WAIT();
    __syncthreads();

    for (int st = 0; st < 98; ++st) {
        const int buf = st & 1;
        if (st + 1 < 98) load_stage(st + 1, buf ^ 1);
        #pragma unroll
        for (int kc = 0; kc < 2; ++kc) {
            const int ac = (kc << 5) + kb8;
            bfx8 alo = *(const bfx8*)&As[buf][arow * PITCH + ac];
            bfx8 ahi = *(const bfx8*)&As[buf][arow * PITCH + ac + 16];
            bfx16 af = __builtin_shufflevector(alo, ahi,
                        0,1,2,3,4,5,6,7,8,9,10,11,12,13,14,15);
            const int bc = (kc << 5) + kb8;
            #pragma unroll
            for (int nf = 0; nf < 8; ++nf) {
                const int px = (nf << 4) + pxl;
                bfx8 blo = *(const bfx8*)&Bs[buf][px * PITCH + bc];
                bfx8 bhi = *(const bfx8*)&Bs[buf][px * PITCH + bc + 16];
                bfx16 bfr = __builtin_shufflevector(blo, bhi,
                            0,1,2,3,4,5,6,7,8,9,10,11,12,13,14,15);
                acc[nf] = __builtin_amdgcn_wmma_f32_16x16x32_bf16(
                    false, af, false, bfr, (short)0, acc[nf], false, false);
            }
        }
        ASYNC_WAIT();
        __syncthreads();
    }

    const int obase = (wid << 4) + ((lane >> 4) << 3);
    #pragma unroll
    for (int nf = 0; nf < 8; ++nf) {
        const int p = n0 + (nf << 4) + pxl;
        #pragma unroll
        for (int j = 0; j < 8; ++j) {
            const int o = obase + j;
            Y[(((size_t)b * 128 + o) << 12) + p] = acc[nf][j];
        }
    }
}

// ---------------------------------------------------------------------------
// Dual 1x1-conv GEMM + fused sigmoid gate:
//   G = sigmoid(W1*X1 + b1 + W2*X2 + b2), WMMA bf16, same async pipeline.
// ---------------------------------------------------------------------------
__global__ __launch_bounds__(256) void k_gemm1x1_gate(
    const bf16* __restrict__ W1, const bf16* __restrict__ X1t,
    const bf16* __restrict__ W2, const bf16* __restrict__ X2t,
    const float* __restrict__ b1, const float* __restrict__ b2,
    float* __restrict__ G)
{
    const int b    = blockIdx.x >> 5;
    const int n0   = (blockIdx.x & 31) << 7;
    const int t    = threadIdx.x;
    const int lane = t & 31, wid = t >> 5;

    __shared__ bf16 As[2][128 * PITCH];
    __shared__ bf16 Bs[2][128 * PITCH];

    const floatx8 zero = {0.f,0.f,0.f,0.f,0.f,0.f,0.f,0.f};
    floatx8 acc[8];
    #pragma unroll
    for (int i = 0; i < 8; ++i) acc[i] = zero;

    const int arow = (wid << 4) + (lane & 15);
    const int kb8  = (lane >> 4) << 3;
    const int kb16 = (lane >> 4) << 4;
    const int pxl  = lane & 15;

    // stage st (0..3): matrix pair = st>>1, 64-channel chunk = st&1
    auto load_stage = [&](int st, int buf) {
        const int m  = st >> 1;
        const int c0 = (st & 1) << 6;
        const bf16* Wm = (m ? W2 : W1) + c0;
        const bf16* Xm = (m ? X2t : X1t) + c0;
        for (int q = t; q < 1024; q += 256) {
            int o = q >> 3, cc = (q & 7) << 3;
            ASYNC_CP16(&As[buf][o * PITCH + cc], &Wm[o * 128 + cc]);
        }
        for (int q = t; q < 1024; q += 256) {
            int px = q >> 3, cc = (q & 7) << 3;
            ASYNC_CP16(&Bs[buf][px * PITCH + cc],
                       &Xm[((((size_t)b << 12) + n0 + px) << 7) + cc]);
        }
    };

    load_stage(0, 0);
    ASYNC_WAIT();
    __syncthreads();

    for (int st = 0; st < 4; ++st) {
        const int buf = st & 1;
        if (st + 1 < 4) load_stage(st + 1, buf ^ 1);
        #pragma unroll
        for (int kc = 0; kc < 2; ++kc) {
            const int ac = (kc << 5) + kb8;
            bfx8 alo = *(const bfx8*)&As[buf][arow * PITCH + ac];
            bfx8 ahi = *(const bfx8*)&As[buf][arow * PITCH + ac + 16];
            bfx16 af = __builtin_shufflevector(alo, ahi,
                        0,1,2,3,4,5,6,7,8,9,10,11,12,13,14,15);
            const int bc = (kc << 5) + kb8;
            #pragma unroll
            for (int nf = 0; nf < 8; ++nf) {
                const int px = (nf << 4) + pxl;
                bfx8 blo = *(const bfx8*)&Bs[buf][px * PITCH + bc];
                bfx8 bhi = *(const bfx8*)&Bs[buf][px * PITCH + bc + 16];
                bfx16 bfr = __builtin_shufflevector(blo, bhi,
                            0,1,2,3,4,5,6,7,8,9,10,11,12,13,14,15);
                acc[nf] = __builtin_amdgcn_wmma_f32_16x16x32_bf16(
                    false, af, false, bfr, (short)0, acc[nf], false, false);
            }
        }
        ASYNC_WAIT();
        __syncthreads();
    }

    const int obase = (wid << 4) + ((lane >> 4) << 3);
    #pragma unroll
    for (int nf = 0; nf < 8; ++nf) {
        const int p = n0 + (nf << 4) + pxl;
        #pragma unroll
        for (int j = 0; j < 8; ++j) {
            const int o = obase + j;
            float v = acc[nf][j] + b1[o] + b2[o];
            G[(((size_t)b * 128 + o) << 12) + p] = 1.f / (1.f + __expf(-v));
        }
    }
}

// ---------------------------------------------------------------------------
// BatchNorm stats per channel (biased var) -> fused scale/bias.
// ---------------------------------------------------------------------------
__global__ __launch_bounds__(256) void k_bnstats(
    const float* __restrict__ X, const float* __restrict__ g,
    const float* __restrict__ bb, float* __restrict__ scale, float* __restrict__ bias)
{
    const int c = blockIdx.x, t = threadIdx.x;
    float s = 0.f, ss = 0.f;
    for (int b = 0; b < NB; ++b) {
        const float* p = X + (((size_t)b * 128 + c) << 12);
        for (int i = t; i < 4096; i += 256) {
            float v = p[i];
            s += v;
            ss = fmaf(v, v, ss);
        }
    }
    __shared__ float rs[256], rss[256];
    rs[t] = s; rss[t] = ss;
    __syncthreads();
    for (int k = 128; k > 0; k >>= 1) {
        if (t < k) { rs[t] += rs[t + k]; rss[t] += rss[t + k]; }
        __syncthreads();
    }
    if (t == 0) {
        const float inv = 1.f / 32768.f;
        float m = rs[0] * inv;
        float v = rss[0] * inv - m * m;
        float r = rsqrtf(v + 1e-3f);
        float sc = g[c] * r;
        scale[c] = sc;
        bias[c] = bb[c] - m * sc;
    }
}

// ---------------------------------------------------------------------------
// Inhibition update (fused BN apply + tanh gating) + NHWC bf16 copy of result.
// ---------------------------------------------------------------------------
__global__ void k_update_inh(
    const float* __restrict__ inp, const float* __restrict__ inh,
    const float* __restrict__ att, const float* __restrict__ raw,
    const float* __restrict__ scale, const float* __restrict__ bias,
    const float* __restrict__ gate, const float* __restrict__ alpha,
    const float* __restrict__ mu, float* __restrict__ out_inh,
    bf16* __restrict__ ninh_bf)
{
    int i = blockIdx.x * 256 + threadIdx.x;
    if (i >= NTOT) return;
    int p = i & 4095;
    int c = (i >> 12) & 127;
    int b = i >> 19;
    float intx = fmaf(raw[i], scale[c], bias[c]);
    float a  = att[((size_t)b << 12) + p];
    float hi = inh[i];
    float gi = hi * a;
    float gx = inp[i] * a;
    float ih = tanhf(gx - tanhf(intx * fmaf(alpha[c], gi, mu[c])));
    float g  = gate[i];
    float ni = (1.f - g) * hi + g * ih;
    out_inh[i] = ni;
    ninh_bf[((((size_t)b << 12) + p) << 7) + c] = (bf16)ni;
}

// ---------------------------------------------------------------------------
// Excitation update (fused BN apply + tanh gating).
// ---------------------------------------------------------------------------
__global__ void k_update_exc(
    const float* __restrict__ exc, const float* __restrict__ ninh,
    const float* __restrict__ raw, const float* __restrict__ scale,
    const float* __restrict__ bias, const float* __restrict__ gate,
    const float* __restrict__ kappa, const float* __restrict__ gam,
    const float* __restrict__ wp, float* __restrict__ out_exc)
{
    int i = blockIdx.x * 256 + threadIdx.x;
    if (i >= NTOT) return;
    int c = (i >> 12) & 127;
    float intx = fmaf(raw[i], scale[c], bias[c]);
    float ni = ninh[i];
    float eh = tanhf(kappa[c] * ni + gam[c] * intx + wp[c] * ni * intx);
    float g  = gate[i];
    out_exc[i] = (1.f - g) * exc[i] + g * eh;
}

// ---------------------------------------------------------------------------
extern "C" void kernel_launch(void* const* d_in, const int* in_sizes, int n_in,
                              void* d_out, int out_size, void* d_ws, size_t ws_size,
                              hipStream_t stream) {
    const float* inp    = (const float*)d_in[0];
    const float* inh    = (const float*)d_in[1];
    const float* exc    = (const float*)d_in[2];
    const float* a_w_w  = (const float*)d_in[3];
    const float* a_w_b  = (const float*)d_in[4];
    const float* a_u_w  = (const float*)d_in[5];
    const float* a_u_b  = (const float*)d_in[6];
    const float* i_w_w  = (const float*)d_in[7];
    const float* i_w_b  = (const float*)d_in[8];
    const float* i_u_w  = (const float*)d_in[9];
    const float* i_u_b  = (const float*)d_in[10];
    const float* e_w_w  = (const float*)d_in[11];
    const float* e_w_b  = (const float*)d_in[12];
    const float* e_u_w  = (const float*)d_in[13];
    const float* e_u_b  = (const float*)d_in[14];
    const float* w_inh  = (const float*)d_in[15];
    const float* w_exc  = (const float*)d_in[16];
    const float* alpha  = (const float*)d_in[17];
    const float* gamma_ = (const float*)d_in[18];
    const float* kappa  = (const float*)d_in[19];
    const float* w_prm  = (const float*)d_in[20];
    const float* mu     = (const float*)d_in[21];
    const float* bn0_g  = (const float*)d_in[22];
    const float* bn0_b  = (const float*)d_in[23];
    const float* bn1_g  = (const float*)d_in[24];
    const float* bn1_b  = (const float*)d_in[25];

    char* ws = (char*)d_ws;
    bf16*  wpk_inh = (bf16*)(ws + 0);
    bf16*  wpk_exc = (bf16*)(ws + 1605632);
    bf16*  wiw     = (bf16*)(ws + 3211264);
    bf16*  wiu     = (bf16*)(ws + 3244032);
    bf16*  wew     = (bf16*)(ws + 3276800);
    bf16*  weu     = (bf16*)(ws + 3309568);
    float* att     = (float*)(ws + 3342336);
    float* bns     = (float*)(ws + 3473408);
    float* bnb     = (float*)(ws + 3473920);
    bf16*  xg_bf   = (bf16*)(ws + 3474432);     // reused as ninh_bf later
    bf16*  inp_bf  = (bf16*)(ws + 11863040);
    bf16*  inh_bf  = (bf16*)(ws + 20251648);
    bf16*  exc_bf  = (bf16*)(ws + 28640256);
    float* Fraw    = (float*)(ws + 37028864);   // conv outputs (reused)
    float* Fgate   = (float*)(ws + 53806080);   // gates (reused)

    float* out_inh = (float*)d_out;
    float* out_exc = out_inh + NTOT;

    // weight preparation
    k_pack_wconv<<<3136, 256, 0, stream>>>(w_inh, wpk_inh);
    k_pack_wconv<<<3136, 256, 0, stream>>>(w_exc, wpk_exc);
    k_f2bf<<<64, 256, 0, stream>>>(i_w_w, wiw, 16384);
    k_f2bf<<<64, 256, 0, stream>>>(i_u_w, wiu, 16384);
    k_f2bf<<<64, 256, 0, stream>>>(e_w_w, wew, 16384);
    k_f2bf<<<64, 256, 0, stream>>>(e_u_w, weu, 16384);

    // attention gate + activation packing
    k_attgate<<<128, 256, 0, stream>>>(inp, exc, a_w_w, a_u_w, a_w_b, a_u_b, att);
    k_pack_acts<<<NTOT / 256, 256, 0, stream>>>(inp, inh, exc, att,
                                                inp_bf, inh_bf, exc_bf, xg_bf);

    // inhibition path
    k_conv7_wmma<<<256, 256, 0, stream>>>(xg_bf, wpk_inh, Fraw);
    k_gemm1x1_gate<<<256, 256, 0, stream>>>(wiw, inp_bf, wiu, inh_bf, i_w_b, i_u_b, Fgate);
    k_bnstats<<<128, 256, 0, stream>>>(Fraw, bn0_g, bn0_b, bns, bnb);
    k_update_inh<<<NTOT / 256, 256, 0, stream>>>(inp, inh, att, Fraw, bns, bnb, Fgate,
                                                 alpha, mu, out_inh, xg_bf);

    // excitation path (xg_bf now holds new inhibition in NHWC bf16)
    k_gemm1x1_gate<<<256, 256, 0, stream>>>(wew, xg_bf, weu, exc_bf, e_w_b, e_u_b, Fgate);
    k_conv7_wmma<<<256, 256, 0, stream>>>(xg_bf, wpk_exc, Fraw);
    k_bnstats<<<128, 256, 0, stream>>>(Fraw, bn1_g, bn1_b, bns, bnb);
    k_update_exc<<<NTOT / 256, 256, 0, stream>>>(exc, out_inh, Fraw, bns, bnb, Fgate,
                                                 kappa, gamma_, w_prm, out_exc);
}
